// GroupedQueryAttention_5403068859031
// MI455X (gfx1250) — compile-verified, matrix operand order checked by reading, not run on hardware
//
#include <hip/hip_runtime.h>

#define DIMN   1024
#define HEADS  16
#define GROUPS 4
#define HD     64
#define SEQ    2048
#define SCALEF 0.125f   // 1/sqrt(64)

typedef unsigned short u16;
typedef __attribute__((ext_vector_type(16))) __bf16 bf16x16;
typedef __attribute__((ext_vector_type(8)))  __bf16 bf16x8;
typedef __attribute__((ext_vector_type(4)))  __bf16 bf16x4;
typedef __attribute__((ext_vector_type(8)))  float  f32x8;
typedef __attribute__((ext_vector_type(4)))  float  f32x4;
typedef __attribute__((ext_vector_type(16))) u16    u16x16;
typedef __attribute__((ext_vector_type(8)))  u16    u16x8;
typedef __attribute__((ext_vector_type(4)))  u16    u16x4;
typedef __attribute__((ext_vector_type(4)))  int    i32x4;

#if defined(__has_builtin)
# if __has_builtin(__builtin_amdgcn_global_load_async_to_lds_b128)
#  define HAVE_ASYNC_LDS 1
# endif
#endif

#ifdef HAVE_ASYNC_LDS
typedef __attribute__((address_space(1))) i32x4* gl_i32x4p;  // global int4*
typedef __attribute__((address_space(3))) i32x4* ld_i32x4p;  // LDS int4*
#endif

__device__ __forceinline__ u16 f2bf(float f) {
  return __builtin_bit_cast(u16, (__bf16)f);   // native cvt, not bit-twiddle
}

__device__ __forceinline__ f32x8 wmma_bf16(bf16x16 a, bf16x16 b, f32x8 c) {
  return __builtin_amdgcn_wmma_f32_16x16x32_bf16(false, a, false, b, (short)0, c,
                                                 false, false);
}

// A/B fragment (16x32 bf16) from bf16(u16) row-major storage (global or LDS).
// rowp points at this lane's row. K slice per ISA layout:
// elems 0..7 -> k0 + half*8, elems 8..15 -> k0 + 16 + half*8.  Two 16B loads.
__device__ __forceinline__ bf16x16 frag_from_bf16(const u16* rowp, int k0) {
  const int lh = (threadIdx.x >> 4) & 1;
  const u16* p = rowp + k0 + lh * 8;
  u16x8 a = *(const u16x8*)(p);
  u16x8 b = *(const u16x8*)(p + 16);
  u16x16 r = __builtin_shufflevector(a, b, 0, 1, 2, 3, 4, 5, 6, 7,
                                     8, 9, 10, 11, 12, 13, 14, 15);
  return __builtin_bit_cast(bf16x16, r);
}

// ---------------------------------------------------------------------------
// Bulk f32 -> bf16 conversion (one streaming pass; native packed cvt)
// ---------------------------------------------------------------------------
__global__ void __launch_bounds__(256)
k_cvt(const float* __restrict__ in, u16* __restrict__ out, int n) {
  int i = (blockIdx.x * 256 + threadIdx.x) * 8;
  if (i + 8 <= n) {
    f32x4 a = *(const f32x4*)(in + i);
    f32x4 b = *(const f32x4*)(in + i + 4);
    u16x4 ra = __builtin_bit_cast(u16x4, __builtin_convertvector(a, bf16x4));
    u16x4 rb = __builtin_bit_cast(u16x4, __builtin_convertvector(b, bf16x4));
    *(u16x8*)(out + i) = __builtin_shufflevector(ra, rb, 0, 1, 2, 3, 4, 5, 6, 7);
  }
}

// ---------------------------------------------------------------------------
// Projection + per-head LayerNorm -> bf16 [B, nslice, SEQ, 64]
// One wave computes a 16-row x 64-col tile (one head slice), K = 1024.
// ---------------------------------------------------------------------------
__global__ void __launch_bounds__(256)
k_proj_ln(const u16* __restrict__ x, const u16* __restrict__ W,
          const float* __restrict__ bias, const float* __restrict__ lnw,
          const float* __restrict__ lnb, u16* __restrict__ out, int nslice) {
  const int lane = threadIdx.x & 31, wv = threadIdx.x >> 5;
  const int sub = lane & 15, lh = lane >> 4;
  const int tile = blockIdx.x * 8 + wv;
  const int rowTile = tile / nslice;
  const int hs = tile % nslice;
  const int r0 = rowTile * 16;
  const u16* arow = x + (size_t)(r0 + sub) * DIMN;
  f32x8 acc[4] = {};
  for (int k0 = 0; k0 < DIMN; k0 += 32) {
    bf16x16 a = frag_from_bf16(arow, k0);
#pragma unroll
    for (int t = 0; t < 4; ++t) {
      const u16* brow = W + (size_t)(hs * 64 + t * 16 + sub) * DIMN;
      acc[t] = wmma_bf16(a, frag_from_bf16(brow, k0), acc[t]);
    }
  }
#pragma unroll
  for (int t = 0; t < 4; ++t) {
    float bv = bias[hs * 64 + t * 16 + sub];
#pragma unroll
    for (int i = 0; i < 8; ++i) acc[t][i] += bv;
  }
  // LayerNorm over 64 columns of each row (rows live in 16-lane halves).
  float mu[8], ri[8];
#pragma unroll
  for (int i = 0; i < 8; ++i) {
    float s = acc[0][i] + acc[1][i] + acc[2][i] + acc[3][i];
    float q = acc[0][i] * acc[0][i] + acc[1][i] * acc[1][i] +
              acc[2][i] * acc[2][i] + acc[3][i] * acc[3][i];
#pragma unroll
    for (int m = 1; m < 16; m <<= 1) {
      s += __shfl_xor(s, m, 32);
      q += __shfl_xor(q, m, 32);
    }
    float mean = s * (1.0f / 64.0f);
    float var  = q * (1.0f / 64.0f) - mean * mean;
    mu[i] = mean;
    ri[i] = rsqrtf(var + 1e-5f);
  }
#pragma unroll
  for (int t = 0; t < 4; ++t) {
    float w = lnw[t * 16 + sub], bb = lnb[t * 16 + sub];
#pragma unroll
    for (int i = 0; i < 8; ++i) {
      int r = r0 + lh * 8 + i;
      int bi = r >> 11, si = r & (SEQ - 1);
      float v = (acc[t][i] - mu[i]) * ri[i] * w + bb;
      out[(((size_t)(bi * nslice + hs)) * SEQ + si) * HD + t * 16 + sub] = f2bf(v);
    }
  }
}

// ---------------------------------------------------------------------------
// V projection, no LN, stored transposed: bf16 [B, G, 64, SEQ]
// ---------------------------------------------------------------------------
__global__ void __launch_bounds__(256)
k_proj_vT(const u16* __restrict__ x, const u16* __restrict__ W,
          const float* __restrict__ bias, u16* __restrict__ out) {
  const int lane = threadIdx.x & 31, wv = threadIdx.x >> 5;
  const int sub = lane & 15, lh = lane >> 4;
  const int tile = blockIdx.x * 8 + wv;
  const int rowTile = tile / GROUPS;
  const int g = tile % GROUPS;
  const int r0 = rowTile * 16;
  const u16* arow = x + (size_t)(r0 + sub) * DIMN;
  f32x8 acc[4] = {};
  for (int k0 = 0; k0 < DIMN; k0 += 32) {
    bf16x16 a = frag_from_bf16(arow, k0);
#pragma unroll
    for (int t = 0; t < 4; ++t) {
      const u16* brow = W + (size_t)(g * 64 + t * 16 + sub) * DIMN;
      acc[t] = wmma_bf16(a, frag_from_bf16(brow, k0), acc[t]);
    }
  }
  const int r = r0 + lh * 8;
  const int bi = r >> 11, kv = r & (SEQ - 1);
#pragma unroll
  for (int t = 0; t < 4; ++t) {
    float bv = bias[g * 64 + t * 16 + sub];
    u16x8 pk;
#pragma unroll
    for (int i = 0; i < 8; ++i) pk[i] = f2bf(acc[t][i] + bv);
    // 8 consecutive kv values per lane -> one 16B store
    *(u16x8*)&out[(((size_t)(bi * GROUPS + g)) * HD + t * 16 + sub) * SEQ + kv] = pk;
  }
}

// ---------------------------------------------------------------------------
// Flash attention: one wave = 16 q-rows x (head_dim=64), 8 waves/WG.
// grid: blockIdx.x = (b*16+h)*16 + qblock
// ---------------------------------------------------------------------------
__global__ void __launch_bounds__(256)
k_attn(const u16* __restrict__ qbf, const u16* __restrict__ kbf,
       const u16* __restrict__ vbfT, const unsigned char* __restrict__ mask,
       u16* __restrict__ obf) {
  __shared__ __align__(16) u16 kT[32 * 64];      // K chunk [32 keys][64 d]
  __shared__ __align__(16) u16 vT[64 * 32];      // V chunk transposed [64 d][32 keys]
  __shared__ __align__(16) u16 pT[8][16 * 32];   // per-wave P tile
  const int tid = threadIdx.x, lane = tid & 31, wv = tid >> 5;
  const int sub = lane & 15, lh = lane >> 4;
  const int qb = blockIdx.x & 15;
  const int bh = blockIdx.x >> 4;                // b*16 + h
  const int bidx = bh >> 4, h = bh & 15, g = h >> 2;
  const int q0 = qb * 128 + wv * 16;

  const u16* qrow = qbf + ((size_t)bh * SEQ + q0 + sub) * HD;
  bf16x16 qa0 = frag_from_bf16(qrow, 0);
  bf16x16 qa1 = frag_from_bf16(qrow, 32);

  const u16* kbase = kbf + (size_t)(bidx * GROUPS + g) * SEQ * HD;
  const u16* vbase = vbfT + (size_t)(bidx * GROUPS + g) * HD * SEQ;
  const unsigned char* mrow = mask + ((size_t)bidx * SEQ + q0) * SEQ;

  f32x8 acc[4] = {};
  float run_m[8], run_l[8];
#pragma unroll
  for (int i = 0; i < 8; ++i) { run_m[i] = -1e30f; run_l[i] = 0.0f; }

  const int kr = tid >> 3, kc8 = (tid & 7) * 8;  // K staging slot
  const int vr = tid >> 2, vc8 = (tid & 3) * 8;  // V staging slot

  for (int kc = 0; kc < SEQ; kc += 32) {
    __syncthreads();
#ifdef HAVE_ASYNC_LDS
    // CDNA5 async global->LDS (bypasses VGPR roundtrip; tracked by ASYNCcnt).
    // Generic-pointer low 32 bits are the LDS byte offset (ISA 10.2 aperture).
    __builtin_amdgcn_global_load_async_to_lds_b128(
        (gl_i32x4p)(size_t)&kbase[(size_t)(kc + kr) * HD + kc8],
        (ld_i32x4p)(unsigned int)(size_t)&kT[kr * 64 + kc8], 0, 0);
    __builtin_amdgcn_global_load_async_to_lds_b128(
        (gl_i32x4p)(size_t)&vbase[(size_t)vr * SEQ + kc + vc8],
        (ld_i32x4p)(unsigned int)(size_t)&vT[vr * 32 + vc8], 0, 0);
    asm volatile("s_wait_asynccnt 0" ::: "memory");
#else
    *(u16x8*)&kT[kr * 64 + kc8] = *(const u16x8*)&kbase[(size_t)(kc + kr) * HD + kc8];
    *(u16x8*)&vT[vr * 32 + vc8] = *(const u16x8*)&vbase[(size_t)vr * SEQ + kc + vc8];
#endif
    __syncthreads();
    if (kc + 32 < SEQ) {
      __builtin_prefetch(&kbase[(size_t)(kc + 32 + kr) * HD + kc8], 0, 3);
      __builtin_prefetch(&vbase[(size_t)vr * SEQ + kc + 32 + vc8], 0, 3);
    }
    // S = Q @ K^T   (16 x 32 scores as two 16x16 fragments)
    f32x8 s[2];
#pragma unroll
    for (int t = 0; t < 2; ++t) {
      const u16* krow = &kT[(t * 16 + sub) * 64];
      f32x8 c = {};
      c = wmma_bf16(qa0, frag_from_bf16(krow, 0), c);
      c = wmma_bf16(qa1, frag_from_bf16(krow, 32), c);
      s[t] = c;
    }
    // scale + mask
#pragma unroll
    for (int t = 0; t < 2; ++t) {
#pragma unroll
      for (int i = 0; i < 8; ++i) {
        float v = s[t][i] * SCALEF;
        const int kv = kc + t * 16 + sub;
        if (!mrow[(size_t)(lh * 8 + i) * SEQ + kv]) v = -3.0e38f;
        s[t][i] = v;
      }
    }
    // online softmax (row stats reduce across 16-lane halves)
    float al[8];
#pragma unroll
    for (int i = 0; i < 8; ++i) {
      float mi = fmaxf(s[0][i], s[1][i]);
#pragma unroll
      for (int m = 1; m < 16; m <<= 1) mi = fmaxf(mi, __shfl_xor(mi, m, 32));
      float nm = fmaxf(run_m[i], mi);
      float a  = __expf(run_m[i] - nm);
      float p0 = __expf(s[0][i] - nm);
      float p1 = __expf(s[1][i] - nm);
      s[0][i] = p0; s[1][i] = p1;
      float rs = p0 + p1;
#pragma unroll
      for (int m = 1; m < 16; m <<= 1) rs += __shfl_xor(rs, m, 32);
      run_l[i] = run_l[i] * a + rs;
      run_m[i] = nm;
      al[i] = a;
    }
#pragma unroll
    for (int t = 0; t < 4; ++t)
#pragma unroll
      for (int i = 0; i < 8; ++i) acc[t][i] *= al[i];
    // P -> LDS (bf16), then reload as WMMA A fragment
#pragma unroll
    for (int t = 0; t < 2; ++t)
#pragma unroll
      for (int i = 0; i < 8; ++i)
        pT[wv][(lh * 8 + i) * 32 + t * 16 + sub] = f2bf(s[t][i]);
    asm volatile("s_wait_dscnt 0" ::: "memory");   // per-wave DS ordering
    bf16x16 pa = frag_from_bf16(&pT[wv][sub * 32], 0);
#pragma unroll
    for (int t = 0; t < 4; ++t)
      acc[t] = wmma_bf16(pa, frag_from_bf16(&vT[(t * 16 + sub) * 32], 0), acc[t]);
  }
  // finalize: divide by row sum, store bf16 [B, Q, H, 64] for the O projection
#pragma unroll
  for (int i = 0; i < 8; ++i) run_l[i] = 1.0f / run_l[i];
#pragma unroll
  for (int t = 0; t < 4; ++t)
#pragma unroll
    for (int i = 0; i < 8; ++i) {
      int r = q0 + lh * 8 + i;
      obf[((size_t)bidx * SEQ + r) * DIMN + h * HD + t * 16 + sub] =
          f2bf(acc[t][i] * run_l[i]);
    }
}

// ---------------------------------------------------------------------------
// Output projection: f32 out = attn_bf16 @ Wo^T + bo
// ---------------------------------------------------------------------------
__global__ void __launch_bounds__(256)
k_oproj(const u16* __restrict__ abf, const u16* __restrict__ W,
        const float* __restrict__ bias, float* __restrict__ out) {
  const int lane = threadIdx.x & 31, wv = threadIdx.x >> 5;
  const int sub = lane & 15, lh = lane >> 4;
  const int tile = blockIdx.x * 8 + wv;   // 4096 tiles
  const int rowTile = tile >> 4;
  const int ct = tile & 15;
  const int r0 = rowTile * 16;
  const u16* arow = abf + (size_t)(r0 + sub) * DIMN;
  f32x8 acc[4] = {};
  for (int k0 = 0; k0 < DIMN; k0 += 32) {
    bf16x16 a = frag_from_bf16(arow, k0);
#pragma unroll
    for (int t = 0; t < 4; ++t) {
      const u16* brow = W + (size_t)(ct * 64 + t * 16 + sub) * DIMN;
      acc[t] = wmma_bf16(a, frag_from_bf16(brow, k0), acc[t]);
    }
  }
#pragma unroll
  for (int t = 0; t < 4; ++t) {
    int col = ct * 64 + t * 16 + sub;
    float bv = bias[col];
#pragma unroll
    for (int i = 0; i < 8; ++i) {
      int r = r0 + lh * 8 + i;
      out[(size_t)r * DIMN + col] = acc[t][i] + bv;
    }
  }
}

extern "C" void kernel_launch(void* const* d_in, const int* in_sizes, int n_in,
                              void* d_out, int out_size, void* d_ws, size_t ws_size,
                              hipStream_t stream) {
  (void)in_sizes; (void)n_in; (void)out_size; (void)ws_size;
  const float* query = (const float*)d_in[0];
  const float* key   = (const float*)d_in[1];
  const float* value = (const float*)d_in[2];
  const unsigned char* mask = (const unsigned char*)d_in[3];
  const float* Wq = (const float*)d_in[4];
  const float* bq = (const float*)d_in[5];
  const float* Wk = (const float*)d_in[6];
  const float* bk = (const float*)d_in[7];
  const float* Wv = (const float*)d_in[8];
  const float* bv = (const float*)d_in[9];
  const float* qnw = (const float*)d_in[10];
  const float* qnb = (const float*)d_in[11];
  const float* knw = (const float*)d_in[12];
  const float* knb = (const float*)d_in[13];
  const float* Wo = (const float*)d_in[14];
  const float* bo = (const float*)d_in[15];
  float* out = (float*)d_out;

  const size_t MB = (size_t)1 << 20;
  char* ws = (char*)d_ws;
  u16* xq   = (u16*)(ws);              // 8 MB  query  bf16 [4096,1024]
  u16* xk   = (u16*)(ws + 8 * MB);     // 8 MB  key    bf16
  u16* xv   = (u16*)(ws + 16 * MB);    // 8 MB  value  bf16
  u16* wq   = (u16*)(ws + 24 * MB);    // 2 MB  Wq bf16 [1024,1024]
  u16* wo   = (u16*)(ws + 26 * MB);    // 2 MB  Wo bf16
  u16* wk   = (u16*)(ws + 28 * MB);    // .5 MB Wk bf16 [256,1024]
  u16* wvv  = (u16*)(ws + 28 * MB + 512 * 1024);  // .5 MB Wv bf16
  u16* qbf  = (u16*)(ws + 29 * MB);    // 8 MB  [B,16,SEQ,64]
  u16* kbf  = (u16*)(ws + 37 * MB);    // 2 MB  [B,4,SEQ,64]
  u16* vbfT = (u16*)(ws + 39 * MB);    // 2 MB  [B,4,64,SEQ]
  u16* abf  = (u16*)(ws + 41 * MB);    // 8 MB  [B,SEQ,1024]

  auto cvt = [&](const float* src, u16* dst, int n) {
    k_cvt<<<(n / 8 + 255) / 256, 256, 0, stream>>>(src, dst, n);
  };
  cvt(query, xq, 4096 * 1024);
  cvt(key,   xk, 4096 * 1024);
  cvt(value, xv, 4096 * 1024);
  cvt(Wq, wq, 1024 * 1024);
  cvt(Wk, wk, 256 * 1024);
  cvt(Wv, wvv, 256 * 1024);
  cvt(Wo, wo, 1024 * 1024);

  k_proj_ln<<<512, 256, 0, stream>>>(xq, wq, bq, qnw, qnb, qbf, HEADS);
  k_proj_ln<<<128, 256, 0, stream>>>(xk, wk, bk, knw, knb, kbf, GROUPS);
  k_proj_vT<<<128, 256, 0, stream>>>(xv, wvv, bv, vbfT);
  k_attn<<<512, 256, 0, stream>>>(qbf, kbf, vbfT, mask, abf);
  k_oproj<<<512, 256, 0, stream>>>(abf, wo, bo, out);
}
